// CLSTM_45896020525262
// MI455X (gfx1250) — matrix operand order repaired and checked by reading, not compiled, not run on hardware
//
#include <hip/hip_runtime.h>
#include <hip/hip_bf16.h>

typedef __attribute__((ext_vector_type(16))) __bf16 v16bf;
typedef __attribute__((ext_vector_type(8)))  float  v8f;

#define BB   64      // batch
#define TT   128     // time steps
#define CC   256     // input channels
#define HH   128     // hidden channels
#define NCLS 7
#define NG   4608    // 4*H*9  (gate rows in matricized form)
#define KX0  2304    // 256*9  (layer0 x-part contraction)
#define KH   1152    // 128*9  (h-part contraction / layer1 x-part)

// ---------------------------------------------------------------------------
// D(MxN) = A(MxK,bf16,row-major) * Bw(NxK,bf16,row-major)^T, f32 accumulate,
// bf16 store. Wave tile: 16(M) x 64(N); block = 8 waves = 32(M) x 256(N).
// blockIdx.z selects a batched problem via strides (fwd/bwd dirs).
// ---------------------------------------------------------------------------
__global__ __launch_bounds__(256)
void gemm_bf16_wmma(const __bf16* __restrict__ A, const __bf16* __restrict__ Bw,
                    __bf16* __restrict__ D, int M, int N, int K,
                    long long sA, long long sB, long long sD) {
  const __bf16* Ab = A + (long long)blockIdx.z * sA;
  const __bf16* Bb = Bw + (long long)blockIdx.z * sB;
  __bf16*       Db = D + (long long)blockIdx.z * sD;

  const int lane = threadIdx.x & 31;
  const int wave = threadIdx.x >> 5;
  const int l16  = lane & 15;
  const int half = lane >> 4;
  const int rowTile = blockIdx.y * 32 + (wave & 1) * 16;
  const int colBase = blockIdx.x * 256 + (wave >> 1) * 64;

  v8f acc[4] = {};
  union AF { uint4 u[2]; v16bf v; };
  const __bf16* aRow = Ab + (long long)(rowTile + l16) * K;

  for (int k0 = 0; k0 < K; k0 += 32) {
    AF a;                              // A frag: half0 -> K {0..15}, half1 -> {16..31}
    a.u[0] = *(const uint4*)(aRow + k0 + half * 16);
    a.u[1] = *(const uint4*)(aRow + k0 + half * 16 + 8);
#pragma unroll
    for (int nn = 0; nn < 4; ++nn) {   // B = W^T: 16 contiguous K values per lane
      const __bf16* bP = Bb + (long long)(colBase + nn * 16 + l16) * K + k0 + half * 16;
      v16bf bv = *(const v16bf*)bP;
      acc[nn] = __builtin_amdgcn_wmma_f32_16x16x32_bf16(
          false, a.v, false, bv, (short)0, acc[nn], false, false);
    }
  }
#pragma unroll
  for (int nn = 0; nn < 4; ++nn) {
    const int col = colBase + nn * 16 + l16;
#pragma unroll
    for (int j = 0; j < 8; ++j) {
      const int rr = rowTile + half * 8 + j;   // C/D: VGPR j -> M=j (+8 for hi lane-half)
      Db[(long long)rr * N + col] = (__bf16)acc[nn][j];
    }
  }
}

// ---------------------------------------------------------------------------
// Fully fused recurrent step (one launch per time step, both dirs via grid.z):
//   Zh = h_in * Wh^T computed with WMMA, where each wave's 4 accumulators are
//   the 4 GATES at the same r-slice -> i/f/o/g math entirely in registers.
//   Epilogue adds Zx[t] + bias, updates c, writes h (bf16, double-buffered)
//   and accumulates the bidirectional h-sum.
// Wave tile: 16(b) x [4 gates x 16 r]; block = 8 waves = 32(b) x 64(r).
// grid = (KH/64, BB/32, 2)
// ---------------------------------------------------------------------------
__global__ __launch_bounds__(256)
void lstm_step_fused(const __bf16* __restrict__ h_in, const __bf16* __restrict__ Wh,
                     const __bf16* __restrict__ Zx,
                     const float* __restrict__ bias_f, const float* __restrict__ bias_b,
                     float* __restrict__ cst, __bf16* __restrict__ h_out,
                     float* __restrict__ hsum, int s) {
  const int d = blockIdx.z;
  const __bf16* Ab = h_in + (long long)d * (BB * KH);
  const __bf16* Bb = Wh + (long long)d * ((long long)NG * KH);

  const int lane = threadIdx.x & 31;
  const int wave = threadIdx.x >> 5;
  const int l16  = lane & 15;
  const int half = lane >> 4;
  const int rowTile = blockIdx.y * 32 + (wave & 1) * 16;          // batch rows
  const int rBase   = blockIdx.x * 64 + (wave >> 1) * 16;         // r = hc*9+op slice

  v8f acc[4] = {};
  union AF { uint4 u[2]; v16bf v; };
  const __bf16* aRow = Ab + (long long)(rowTile + l16) * KH;

  for (int k0 = 0; k0 < KH; k0 += 32) {
    AF a;
    a.u[0] = *(const uint4*)(aRow + k0 + half * 16);
    a.u[1] = *(const uint4*)(aRow + k0 + half * 16 + 8);
#pragma unroll
    for (int g = 0; g < 4; ++g) {      // gate g -> W row g*KH + rBase + l16
      const __bf16* bP = Bb + (long long)(g * KH + rBase + l16) * KH + k0 + half * 16;
      v16bf bv = *(const v16bf*)bP;
      acc[g] = __builtin_amdgcn_wmma_f32_16x16x32_bf16(
          false, a.v, false, bv, (short)0, acc[g], false, false);
    }
  }

  // ---- gate epilogue (all four gates for (row, col) live in this lane) ----
  const int t = d ? (TT - 1 - s) : s;
  const float* bias = d ? bias_b : bias_f;
  const int col = rBase + l16;          // r in [0, 1152)
  const int hc  = col / 9;
  const float bi = bias[0 * HH + hc];
  const float bf_ = bias[1 * HH + hc];
  const float bo = bias[2 * HH + hc];
  const float bg = bias[3 * HH + hc];
  const long long zxDir = (long long)d * (8192LL * NG);

#pragma unroll
  for (int j = 0; j < 8; ++j) {
    const int rr = rowTile + half * 8 + j;                       // batch index
    const long long zxB = zxDir + ((long long)t * BB + rr) * NG;
    float zi = acc[0][j] + (float)Zx[zxB + 0 * KH + col] + bi;
    float zf = acc[1][j] + (float)Zx[zxB + 1 * KH + col] + bf_;
    float zo = acc[2][j] + (float)Zx[zxB + 2 * KH + col] + bo;
    float zg = acc[3][j] + (float)Zx[zxB + 3 * KH + col] + bg;
    const long long st = (long long)d * (BB * KH) + (long long)rr * KH + col;
    float c = cst[st];
    float si = 1.f / (1.f + __expf(-zi));
    float sf = 1.f / (1.f + __expf(-zf));
    float so = 1.f / (1.f + __expf(-zo));
    c = sf * c + si * tanhf(zg);
    float h = so * tanhf(c);
    cst[st] = c;
    h_out[st] = (__bf16)h;
    // dirs write different t at a given step (T even) -> race-free accumulate
    hsum[((long long)t * BB + rr) * KH + col] += h;
  }
}

// ---------------------------------------------------------------------------
// Matricize conv weights: Wm[(co*9+op)][(ci*9+ip)] = w[co][cOff+ci][ky][kx]
// with zeros for out-of-range taps (SAME padding on a 3x3 grid).
// ---------------------------------------------------------------------------
__global__ void wprep(const float* __restrict__ w, __bf16* __restrict__ Wm,
                      int CinTot, int cOff, int Cdst) {
  long long n = (long long)NG * Cdst * 9;
  long long e = (long long)blockIdx.x * 256 + threadIdx.x;
  if (e >= n) return;
  int cols = Cdst * 9;
  int row = (int)(e / cols), col = (int)(e % cols);
  int co = row / 9, op = row % 9, oy = op / 3, ox = op % 3;
  int ci = col / 9, ip = col % 9, iy = ip / 3, ix = ip % 3;
  int ky = iy - oy + 1, kx = ix - ox + 1;
  float v = 0.f;
  if (ky >= 0 && ky < 3 && kx >= 0 && kx < 3)
    v = w[(((long long)co * CinTot + cOff + ci) * 3 + ky) * 3 + kx];
  Wm[e] = (__bf16)v;
}

// x (B,T,C,3,3) f32 -> Xbf rows (t*B+b), cols (c*9+p), bf16
__global__ void xconv(const float* __restrict__ x, __bf16* __restrict__ X) {
  long long e = (long long)blockIdx.x * 256 + threadIdx.x;
  int p = (int)(e % 9); long long q = e / 9;
  int c = (int)(q % CC); q /= CC;
  int b = (int)(q % BB); int t = (int)(q / BB);
  X[e] = (__bf16)x[(((long long)b * TT + t) * CC + c) * 9 + p];
}

__global__ void f2bf(const float* __restrict__ s, __bf16* __restrict__ d2) {
  long long e = (long long)blockIdx.x * 256 + threadIdx.x;
  d2[e] = (__bf16)s[e];
}

// out[(b*T+t)*7+cls] = dot(hsum1[t*B+b], fc_w[cls]) + fc_b[cls]
__global__ void fc_kernel(const float* __restrict__ Hs, const float* __restrict__ Wf,
                          const float* __restrict__ bf, float* __restrict__ out) {
  int e = blockIdx.x * 256 + threadIdx.x;   // 64*128*7
  int cls = e % NCLS; int q = e / NCLS;
  int t = q % TT; int b = q / TT;
  const float* hr = Hs + ((long long)t * BB + b) * KH;
  const float* wr = Wf + (long long)cls * KH;
  float acc = bf[cls];
  for (int k = 0; k < KH; k += 4)
    acc += hr[k] * wr[k] + hr[k + 1] * wr[k + 1] + hr[k + 2] * wr[k + 2] + hr[k + 3] * wr[k + 3];
  out[e] = acc;
}

extern "C" void kernel_launch(void* const* d_in, const int* in_sizes, int n_in,
                              void* d_out, int out_size, void* d_ws, size_t ws_size,
                              hipStream_t stream) {
  const float* x    = (const float*)d_in[0];
  const float* w_f0 = (const float*)d_in[1];
  const float* b_f0 = (const float*)d_in[2];
  const float* w_b0 = (const float*)d_in[3];
  const float* b_b0 = (const float*)d_in[4];
  const float* w_f1 = (const float*)d_in[5];
  const float* b_f1 = (const float*)d_in[6];
  const float* w_b1 = (const float*)d_in[7];
  const float* b_b1 = (const float*)d_in[8];
  const float* fc_w = (const float*)d_in[9];
  const float* fc_b = (const float*)d_in[10];
  float* out = (float*)d_out;
  (void)in_sizes; (void)n_in; (void)out_size; (void)ws_size;

  // workspace carve (256B aligned)
  char* p = (char*)d_ws;
  auto carve = [&](size_t bytes) { char* r = p; p += (bytes + 255) & ~(size_t)255; return r; };
  __bf16* Xbf0 = (__bf16*)carve(8192LL * KX0 * 2);        // x-part inputs, layer0
  __bf16* Xbf1 = (__bf16*)carve(8192LL * KH * 2);         // layer1 inputs (= hsum0 bf16)
  __bf16* Wx0  = (__bf16*)carve(2LL * NG * KX0 * 2);      // [dir] layer0 x-weights
  __bf16* Wx1  = (__bf16*)carve(2LL * NG * KH * 2);       // [dir] layer1 x-weights
  __bf16* Wh0  = (__bf16*)carve(2LL * NG * KH * 2);       // [dir] layer0 h-weights
  __bf16* Wh1  = (__bf16*)carve(2LL * NG * KH * 2);       // [dir] layer1 h-weights
  __bf16* Zx   = (__bf16*)carve(2LL * 8192 * NG * 2);     // [dir] precomputed x-contributions
  __bf16* hA   = (__bf16*)carve(2LL * BB * KH * 2);       // [dir] hidden state ping
  __bf16* hB   = (__bf16*)carve(2LL * BB * KH * 2);       // [dir] hidden state pong
  float*  cst  = (float*)carve(2LL * BB * KH * 4);        // [dir] cell state
  float*  hs0  = (float*)carve(8192LL * KH * 4);          // layer0 fwd+bwd h sum
  float*  hs1  = (float*)carve(8192LL * KH * 4);          // layer1 fwd+bwd h sum

  hipMemsetAsync(hA, 0, 2LL * BB * KH * 2, stream);
  hipMemsetAsync(hB, 0, 2LL * BB * KH * 2, stream);
  hipMemsetAsync(cst, 0, 2LL * BB * KH * 4, stream);
  hipMemsetAsync(hs0, 0, 8192LL * KH * 4, stream);
  hipMemsetAsync(hs1, 0, 8192LL * KH * 4, stream);

  // weight matricization (x-part / h-part split of each conv weight)
  auto gw = [](long long n) { return dim3((unsigned)((n + 255) / 256)); };
  wprep<<<gw((long long)NG * KX0), 256, 0, stream>>>(w_f0, Wx0,                        384, 0,   256);
  wprep<<<gw((long long)NG * KX0), 256, 0, stream>>>(w_b0, Wx0 + (long long)NG * KX0,  384, 0,   256);
  wprep<<<gw((long long)NG * KH ), 256, 0, stream>>>(w_f0, Wh0,                        384, 256, 128);
  wprep<<<gw((long long)NG * KH ), 256, 0, stream>>>(w_b0, Wh0 + (long long)NG * KH,   384, 256, 128);
  wprep<<<gw((long long)NG * KH ), 256, 0, stream>>>(w_f1, Wx1,                        256, 0,   128);
  wprep<<<gw((long long)NG * KH ), 256, 0, stream>>>(w_b1, Wx1 + (long long)NG * KH,   256, 0,   128);
  wprep<<<gw((long long)NG * KH ), 256, 0, stream>>>(w_f1, Wh1,                        256, 128, 128);
  wprep<<<gw((long long)NG * KH ), 256, 0, stream>>>(w_b1, Wh1 + (long long)NG * KH,   256, 128, 128);

  xconv<<<dim3(73728), 256, 0, stream>>>(x, Xbf0);

  // ---- layer 0: time-parallel x-contribution GEMM (both dirs), then recurrence
  gemm_bf16_wmma<<<dim3(NG / 256, 8192 / 32, 2), 256, 0, stream>>>(
      Xbf0, Wx0, Zx, 8192, NG, KX0, 0LL, (long long)NG * KX0, 8192LL * NG);

  for (int s = 0; s < TT; ++s) {
    const __bf16* hin = (s & 1) ? hB : hA;
    __bf16*      hout = (s & 1) ? hA : hB;
    lstm_step_fused<<<dim3(KH / 64, BB / 32, 2), 256, 0, stream>>>(
        hin, Wh0, Zx, b_f0, b_b0, cst, hout, hs0, s);
  }

  // ---- layer transition
  hipMemsetAsync(hA, 0, 2LL * BB * KH * 2, stream);
  hipMemsetAsync(hB, 0, 2LL * BB * KH * 2, stream);
  hipMemsetAsync(cst, 0, 2LL * BB * KH * 4, stream);
  f2bf<<<dim3(36864), 256, 0, stream>>>(hs0, Xbf1);

  // ---- layer 1
  gemm_bf16_wmma<<<dim3(NG / 256, 8192 / 32, 2), 256, 0, stream>>>(
      Xbf1, Wx1, Zx, 8192, NG, KH, 0LL, (long long)NG * KH, 8192LL * NG);

  for (int s = 0; s < TT; ++s) {
    const __bf16* hin = (s & 1) ? hB : hA;
    __bf16*      hout = (s & 1) ? hA : hB;
    lstm_step_fused<<<dim3(KH / 64, BB / 32, 2), 256, 0, stream>>>(
        hin, Wh1, Zx, b_f1, b_b1, cst, hout, hs1, s);
  }

  fc_kernel<<<dim3(224), 256, 0, stream>>>(hs1, fc_w, fc_b, out);
}
